// CustomGRUCell_3435973837372
// MI455X (gfx1250) — matrix operand order repaired and checked by reading, not compiled, not run on hardware
//
#include <hip/hip_runtime.h>

// ---------------------------------------------------------------------------
// GRU cell for MI455X (gfx1250, wave32, WMMA + async global->LDS copies).
// bf16 inputs to v_wmma_f32_16x16x32_bf16, fp32 accumulation and epilogues.
// ---------------------------------------------------------------------------

typedef __attribute__((ext_vector_type(16))) __bf16 v16bf;
typedef __attribute__((ext_vector_type(8)))  float  v8f;
typedef __attribute__((ext_vector_type(4)))  int    v4i;
typedef __attribute__((address_space(1))) v4i as1_v4i;   // global int4
typedef __attribute__((address_space(3))) v4i as3_v4i;   // LDS int4

#define GRU_B   8192
#define GRU_IN  1024
#define GRU_H   2048

#define TM 128                 // block tile M
#define TN 256                 // block tile N
#define TK 32                  // K step (one WMMA K)
#define LDSS 40                // padded LDS row stride in bf16 (32 + 8 pad)
#define A_US  (TM * LDSS)      // A tile in ushorts (5120)
#define B_US  (TN * LDSS)      // B tile in ushorts (10240)
#define BUF_US (A_US + B_US)   // one double-buffer slot (15360 ushorts)

#if __has_builtin(__builtin_amdgcn_global_load_async_to_lds_b128)
#define USE_ASYNC 1
#else
#define USE_ASYNC 0
#endif

__device__ __forceinline__ void copy16(const unsigned short* g, unsigned short* l) {
#if USE_ASYNC
    __builtin_amdgcn_global_load_async_to_lds_b128(
        (as1_v4i*)g, (as3_v4i*)l, 0, 0);
#else
    *(uint4*)l = *(const uint4*)g;
#endif
}

__device__ __forceinline__ void wait_async_all() {
#if USE_ASYNC
#if __has_builtin(__builtin_amdgcn_s_wait_asynccnt)
    __builtin_amdgcn_s_wait_asynccnt(0);
#else
    asm volatile("s_wait_asynccnt 0x0" ::: "memory");
#endif
#endif
}

// ------------------------- f32 -> bf16 (RNE) -------------------------------
__global__ void f32_to_bf16_kernel(const float* __restrict__ in,
                                   unsigned short* __restrict__ out,
                                   size_t n) {
    size_t i = (size_t)blockIdx.x * blockDim.x + threadIdx.x;
    size_t stride = (size_t)gridDim.x * blockDim.x;
    for (; i < n; i += stride) {
        unsigned int u = __float_as_uint(in[i]);
        unsigned int r = (u + 0x7FFFu + ((u >> 16) & 1u)) >> 16;
        out[i] = (unsigned short)r;
    }
}

// ------------------------- bf16 WMMA GEMM ----------------------------------
// C[M x N] (f32) = A[M x K] (bf16) * Bw[N x K]^T (bf16), optionally += .
// M % 128 == 0, N % 256 == 0, K % 32 == 0 (always true here).
__global__ __launch_bounds__(256)
void gemm_bf16_wmma(const unsigned short* __restrict__ A,
                    const unsigned short* __restrict__ Bw,
                    float* __restrict__ C,
                    int M, int N, int K, int accumulate) {
    __shared__ unsigned short lds[2 * BUF_US];   // 60 KB

    const int tid  = threadIdx.x;
    const int wave = tid >> 5;
    const int lane = tid & 31;
    const int kg   = lane >> 4;    // half-wave group
    const int r16  = lane & 15;

    const int blockM = blockIdx.y * TM;
    const int blockN = blockIdx.x * TN;
    const int waveM  = (wave >> 2) * 64;    // 0 or 64
    const int waveN  = (wave & 3)  * 64;    // 0,64,128,192

    // ---- per-thread copy chunks (16B each) ----
    // A tile: 128 rows x 64B -> 512 chunks; thread does chunks {tid, tid+256}
    // B tile: 256 rows x 64B -> 1024 chunks; thread does {tid + i*256, i<4}
    const int cRow = tid >> 2;
    const int cSeg = tid & 3;
    const unsigned short* gA = A  + (size_t)(blockM + cRow) * K + cSeg * 8;
    const unsigned short* gB = Bw + (size_t)(blockN + cRow) * K + cSeg * 8;
    const int lOffA = cRow * LDSS + cSeg * 8;
    const int lOffB = A_US + cRow * LDSS + cSeg * 8;

    v8f acc[4][4];
#pragma unroll
    for (int i = 0; i < 4; ++i)
#pragma unroll
        for (int j = 0; j < 4; ++j)
            acc[i][j] = (v8f){0.f, 0.f, 0.f, 0.f, 0.f, 0.f, 0.f, 0.f};

    const int kSteps = K / TK;

    // prologue: tile 0 -> buffer 0
#pragma unroll
    for (int i = 0; i < 2; ++i)
        copy16(gA + (size_t)i * 64 * K, &lds[lOffA + i * 64 * LDSS]);
#pragma unroll
    for (int i = 0; i < 4; ++i)
        copy16(gB + (size_t)i * 64 * K, &lds[lOffB + i * 64 * LDSS]);
    wait_async_all();
    __syncthreads();

    int cur = 0;
    for (int ks = 0; ks < kSteps; ++ks) {
        // issue copies for next tile into the other buffer (consumed 1 iter ago)
        if (ks + 1 < kSteps) {
            const int kb  = (ks + 1) * TK;
            const int nb  = (cur ^ 1) * BUF_US;
#pragma unroll
            for (int i = 0; i < 2; ++i)
                copy16(gA + (size_t)i * 64 * K + kb, &lds[nb + lOffA + i * 64 * LDSS]);
#pragma unroll
            for (int i = 0; i < 4; ++i)
                copy16(gB + (size_t)i * 64 * K + kb, &lds[nb + lOffB + i * 64 * LDSS]);
        }

        const unsigned short* As = &lds[cur * BUF_US];
        const unsigned short* Bs = &lds[cur * BUF_US + A_US];

        // A fragments: lane group 0 holds K {0..7,16..23}, group 1 {8..15,24..31}
        v16bf aFrag[4];
#pragma unroll
        for (int i = 0; i < 4; ++i) {
            const unsigned short* p = As + (waveM + i * 16 + r16) * LDSS;
            union { uint4 u[2]; v16bf v; } cv;
            cv.u[0] = *(const uint4*)(p + kg * 8);
            cv.u[1] = *(const uint4*)(p + 16 + kg * 8);
            aFrag[i] = cv.v;
        }
        // B fragments: lane group 0 holds K {0..7,16..23}, group 1 {8..15,24..31}
#pragma unroll
        for (int j = 0; j < 4; ++j) {
            const unsigned short* p = Bs + (waveN + j * 16 + r16) * LDSS;
            union { uint4 u[2]; v16bf v; } cv;
            cv.u[0] = *(const uint4*)(p + kg * 8);
            cv.u[1] = *(const uint4*)(p + 16 + kg * 8);
            v16bf bFrag = cv.v;
#pragma unroll
            for (int i = 0; i < 4; ++i)
                acc[i][j] = __builtin_amdgcn_wmma_f32_16x16x32_bf16(
                    false, aFrag[i], false, bFrag, (short)0, acc[i][j], false, false);
        }

        wait_async_all();     // own async copies for the other buffer complete
        __syncthreads();      // all waves done reading `cur`, all copies visible
        cur ^= 1;
    }

    // Epilogue: C/D layout -> lane n = lane&15, VGPR v -> M = v + 8*(lane>=16)
#pragma unroll
    for (int i = 0; i < 4; ++i)
#pragma unroll
        for (int j = 0; j < 4; ++j) {
            int baseM = blockM + waveM + i * 16 + kg * 8;
            int baseN = blockN + waveN + j * 16 + r16;
            float* cp = C + (size_t)baseM * N + baseN;
#pragma unroll
            for (int v = 0; v < 8; ++v) {
                float* q = cp + (size_t)v * N;
                float val = acc[i][j][v];
                if (accumulate) *q = *q + val; else *q = val;
            }
        }
}

// ------------------------- gate kernel -------------------------------------
// gr holds gi_r + gh_r (f32); gz holds gi_z + gh_z and is overwritten with z.
// rh <- bf16( sigmoid(gr + b_r) * h_prev )
__global__ void gru_gates_kernel(const float* __restrict__ gr,
                                 float* __restrict__ gz_inout,
                                 const float* __restrict__ h,
                                 const float* __restrict__ br,
                                 const float* __restrict__ bz,
                                 unsigned short* __restrict__ rh,
                                 size_t n) {
    size_t i = (size_t)blockIdx.x * blockDim.x + threadIdx.x;
    size_t stride = (size_t)gridDim.x * blockDim.x;
    for (; i < n; i += stride) {
        int col = (int)(i & (GRU_H - 1));
        float r = 1.0f / (1.0f + __expf(-(gr[i] + br[col])));
        float z = 1.0f / (1.0f + __expf(-(gz_inout[i] + bz[col])));
        gz_inout[i] = z;
        float v = r * h[i];
        unsigned int u = __float_as_uint(v);
        rh[i] = (unsigned short)((u + 0x7FFFu + ((u >> 16) & 1u)) >> 16);
    }
}

// ------------------------- output kernel -----------------------------------
// gh_acc holds gi_h + (r*h)@W_hh^T.  out = (1-z)*h + z*tanh(gh_acc + b_h)
__global__ void gru_out_kernel(const float* __restrict__ gh_acc,
                               const float* __restrict__ z,
                               const float* __restrict__ h,
                               const float* __restrict__ bh,
                               float* __restrict__ out,
                               size_t n) {
    size_t i = (size_t)blockIdx.x * blockDim.x + threadIdx.x;
    size_t stride = (size_t)gridDim.x * blockDim.x;
    for (; i < n; i += stride) {
        int col = (int)(i & (GRU_H - 1));
        float ht = tanhf(gh_acc[i] + bh[col]);
        float zz = z[i];
        out[i] = (1.0f - zz) * h[i] + zz * ht;
    }
}

// ---------------------------------------------------------------------------
extern "C" void kernel_launch(void* const* d_in, const int* in_sizes, int n_in,
                              void* d_out, int out_size, void* d_ws, size_t ws_size,
                              hipStream_t stream) {
    (void)in_sizes; (void)n_in; (void)out_size; (void)ws_size;

    const float* x   = (const float*)d_in[0];
    const float* h   = (const float*)d_in[1];
    const float* Wir = (const float*)d_in[2];
    const float* Wiz = (const float*)d_in[3];
    const float* Wih = (const float*)d_in[4];
    const float* Whr = (const float*)d_in[5];
    const float* Whz = (const float*)d_in[6];
    const float* Whh = (const float*)d_in[7];
    const float* br  = (const float*)d_in[8];
    const float* bz  = (const float*)d_in[9];
    const float* bh  = (const float*)d_in[10];
    float* out = (float*)d_out;

    // ---- workspace layout ----
    char* ws = (char*)d_ws;
    size_t off = 0;
    auto take = [&](size_t bytes) { char* p = ws + off; off += bytes; return p; };

    unsigned short* xb   = (unsigned short*)take((size_t)GRU_B * GRU_IN * 2);
    unsigned short* hb   = (unsigned short*)take((size_t)GRU_B * GRU_H  * 2);
    unsigned short* Wirb = (unsigned short*)take((size_t)GRU_H * GRU_IN * 2);
    unsigned short* Wizb = (unsigned short*)take((size_t)GRU_H * GRU_IN * 2);
    unsigned short* Wihb = (unsigned short*)take((size_t)GRU_H * GRU_IN * 2);
    unsigned short* Whrb = (unsigned short*)take((size_t)GRU_H * GRU_H  * 2);
    unsigned short* Whzb = (unsigned short*)take((size_t)GRU_H * GRU_H  * 2);
    unsigned short* Whhb = (unsigned short*)take((size_t)GRU_H * GRU_H  * 2);
    unsigned short* rhb  = (unsigned short*)take((size_t)GRU_B * GRU_H  * 2);
    float* giR = (float*)take((size_t)GRU_B * GRU_H * 4);
    float* giZ = (float*)take((size_t)GRU_B * GRU_H * 4);
    float* giH = (float*)take((size_t)GRU_B * GRU_H * 4);

    const size_t nBH = (size_t)GRU_B * GRU_H;

    // ---- 1. convert to bf16 ----
    f32_to_bf16_kernel<<<2048, 256, 0, stream>>>(x,   xb,   (size_t)GRU_B * GRU_IN);
    f32_to_bf16_kernel<<<4096, 256, 0, stream>>>(h,   hb,   nBH);
    f32_to_bf16_kernel<<<1024, 256, 0, stream>>>(Wir, Wirb, (size_t)GRU_H * GRU_IN);
    f32_to_bf16_kernel<<<1024, 256, 0, stream>>>(Wiz, Wizb, (size_t)GRU_H * GRU_IN);
    f32_to_bf16_kernel<<<1024, 256, 0, stream>>>(Wih, Wihb, (size_t)GRU_H * GRU_IN);
    f32_to_bf16_kernel<<<2048, 256, 0, stream>>>(Whr, Whrb, (size_t)GRU_H * GRU_H);
    f32_to_bf16_kernel<<<2048, 256, 0, stream>>>(Whz, Whzb, (size_t)GRU_H * GRU_H);
    f32_to_bf16_kernel<<<2048, 256, 0, stream>>>(Whh, Whhb, (size_t)GRU_H * GRU_H);

    dim3 gg(GRU_H / TN, GRU_B / TM);   // (8, 64)

    // ---- 2. input-side projections (beta = 0) ----
    gemm_bf16_wmma<<<gg, 256, 0, stream>>>(xb, Wirb, giR, GRU_B, GRU_H, GRU_IN, 0);
    gemm_bf16_wmma<<<gg, 256, 0, stream>>>(xb, Wizb, giZ, GRU_B, GRU_H, GRU_IN, 0);
    gemm_bf16_wmma<<<gg, 256, 0, stream>>>(xb, Wihb, giH, GRU_B, GRU_H, GRU_IN, 0);

    // ---- 3. hidden-side projections accumulate (beta = 1) ----
    gemm_bf16_wmma<<<gg, 256, 0, stream>>>(hb, Whrb, giR, GRU_B, GRU_H, GRU_H, 1);
    gemm_bf16_wmma<<<gg, 256, 0, stream>>>(hb, Whzb, giZ, GRU_B, GRU_H, GRU_H, 1);

    // ---- 4. gates: r (consumed), z in-place, rh in bf16 ----
    gru_gates_kernel<<<8192, 256, 0, stream>>>(giR, giZ, h, br, bz, rhb, nBH);

    // ---- 5. candidate projection accumulates into giH ----
    gemm_bf16_wmma<<<gg, 256, 0, stream>>>(rhb, Whhb, giH, GRU_B, GRU_H, GRU_H, 1);

    // ---- 6. output ----
    gru_out_kernel<<<8192, 256, 0, stream>>>(giH, giZ, h, bh, out, nBH);
}
